// MultiHeadSelfAttention_6055903887839
// MI455X (gfx1250) — matrix operand order repaired and checked by reading, not compiled, not run on hardware
//
#include <hip/hip_runtime.h>
#include <hip/hip_bf16.h>

typedef __attribute__((ext_vector_type(8)))  _Float16 v8h;
typedef __attribute__((ext_vector_type(16))) _Float16 v16h;
typedef __attribute__((ext_vector_type(8)))  float    v8f;

#define D_MODEL 512
#define SEQ     2048
#define BATCH   2
#define NHEAD   8
#define DK      64
#define NTOK    (BATCH * SEQ)   // 4096

// ---------------------------------------------------------------------------
// Fragment loaders (wave32 WMMA layouts, CDNA5 ISA 7.12.2)
// A (f16, 16x32, MxK): lane<16 row M=l16 holds K 0..7 & 16..23; lane>=16 holds K 8..15 & 24..31
// B (f16, 32x16, KxN): col N = lane&15; lanes 0-15 hold K 0..7 & 16..23, lanes 16-31 hold K 8..15 & 24..31,
//                      contiguous when memory addr = N*ld + K.
// C (f32, 16x16):      VGPR r -> row r + 8*(lane>>4), col = lane&15
// ---------------------------------------------------------------------------
__device__ __forceinline__ v16h load_a_frag(const _Float16* base, int ld, int l16, int half) {
    union { v16h v; v8h h[2]; } u;
    u.h[0] = *(const v8h*)(base + l16 * ld + 8 * half);
    u.h[1] = *(const v8h*)(base + l16 * ld + 16 + 8 * half);
    return u.v;
}

__device__ __forceinline__ v16h load_b_frag(const _Float16* base, int ld, int l16, int half) {
    union { v16h v; v8h h[2]; } u;
    u.h[0] = *(const v8h*)(base + l16 * ld + 8 * half);
    u.h[1] = *(const v8h*)(base + l16 * ld + 16 + 8 * half);
    return u.v;
}

__device__ __forceinline__ v8f wmma16(v16h a, v16h b, v8f c) {
    return __builtin_amdgcn_wmma_f32_16x16x32_f16(false, a, false, b, (short)0, c, false, false);
}

// ---------------------------------------------------------------------------
// fp32 -> fp16 conversion (x and weights)
// ---------------------------------------------------------------------------
__global__ void cvt_f32_f16(const float* __restrict__ s, _Float16* __restrict__ d, int n) {
    int i = blockIdx.x * blockDim.x + threadIdx.x;
    int stride = gridDim.x * blockDim.x;
    for (; i < n; i += stride) d[i] = (_Float16)s[i];
}

// ---------------------------------------------------------------------------
// QKV projection: Y = x @ W^T + b.  One wave computes a 16(token) x 64(e) tile.
// Q,K stored [b,h,s,dk]; V stored transposed [b,h,dk,s] so attention PV
// B-fragments are contiguous per-lane loads.
// ---------------------------------------------------------------------------
__global__ void qkv_proj(const _Float16* __restrict__ xh,
                         const _Float16* __restrict__ wq, const _Float16* __restrict__ wk,
                         const _Float16* __restrict__ wv,
                         const float* __restrict__ bq, const float* __restrict__ bk,
                         const float* __restrict__ bv,
                         _Float16* __restrict__ qh, _Float16* __restrict__ kh,
                         _Float16* __restrict__ vth) {
    const int lane = threadIdx.x & 31;
    const int l16 = lane & 15, half = lane >> 4;
    const int w   = blockIdx.x * (blockDim.x >> 5) + (threadIdx.x >> 5);
    const int mat = w / 2048;        // 0=Q 1=K 2=V
    const int t   = w % 2048;
    const int tm  = t % 256;         // token tile (NTOK/16)
    const int tn  = t / 256;         // e tile of 64

    const _Float16* W    = (mat == 0) ? wq : (mat == 1) ? wk : wv;
    const float*    bias = (mat == 0) ? bq : (mat == 1) ? bk : bv;
    const _Float16* xa   = xh + tm * 16 * D_MODEL;

    v8f c[4] = {};
    for (int kc = 0; kc < D_MODEL / 32; ++kc) {
        v16h a = load_a_frag(xa + kc * 32, D_MODEL, l16, half);
#pragma unroll
        for (int j = 0; j < 4; ++j) {
            v16h b = load_b_frag(W + (tn * 64 + j * 16) * D_MODEL + kc * 32, D_MODEL, l16, half);
            c[j] = wmma16(a, b, c[j]);
        }
    }

    const int t0 = tm * 16;
#pragma unroll
    for (int j = 0; j < 4; ++j) {
        const int e  = tn * 64 + j * 16 + l16;
        const float bb = bias[e];
        const int hI = e >> 6, dk = e & 63;
        if (mat < 2) {
            _Float16* dst = (mat == 0) ? qh : kh;
#pragma unroll
            for (int r = 0; r < 8; ++r) {
                int tok = t0 + r + 8 * half;
                int bI = tok >> 11, s = tok & (SEQ - 1);
                dst[(((bI * NHEAD + hI) * SEQ) + s) * DK + dk] = (_Float16)(c[j][r] + bb);
            }
        } else {
            int bI = t0 >> 11;
            int s0 = (t0 & (SEQ - 1)) + 8 * half;
            v8h pk;
#pragma unroll
            for (int r = 0; r < 8; ++r) pk[r] = (_Float16)(c[j][r] + bb);
            *(v8h*)(vth + (((bI * NHEAD + hI) * DK) + dk) * SEQ + s0) = pk;
        }
    }
}

// ---------------------------------------------------------------------------
// Flash attention with async K/V staging.
// One block = 8 waves, all on the same (b,h), q tiles adjacent, 16 rows apart (8*16=128 rows per block).
// K/V 32-column tiles are cooperatively copied to LDS with
// GLOBAL_LOAD_ASYNC_TO_LDS_B128 (ASYNCcnt), double-buffered, so all 8 waves
// share one copy instead of 8 redundant global streams.
//
// LDS map (halfs): [0,4096)  kbuf[2][32*64]   (K rows s_rel x dk, ld=64)
//                  [4096,8192) vbuf[2][64*32] (V^T rows dk x s_rel, ld=32)
//                  [8192,12288) per-wave 16x32 prob staging
// ---------------------------------------------------------------------------
__global__ void attn_kernel(const _Float16* __restrict__ qh, const _Float16* __restrict__ kh,
                            const _Float16* __restrict__ vth, _Float16* __restrict__ oh) {
    __shared__ _Float16 smem[12288];     // 24 KB
    const int lane = threadIdx.x & 31;
    const int l16 = lane & 15, half = lane >> 4;
    const int wIb = threadIdx.x >> 5;
    const int w   = blockIdx.x * 8 + wIb;
    const int bh  = w >> 7;              // (b*H+h) -- uniform across the block
    const int q0  = (w & 127) * 16;

    const _Float16* qb = qh + (bh * SEQ + q0) * DK;
    const _Float16* kb = kh + bh * SEQ * DK;
    const _Float16* vb = vth + bh * DK * SEQ;
    _Float16* pl = smem + 8192 + wIb * 512;

    // cooperative async stage of one 32-column K/V step into buffer `bufn`
    auto issue_loads = [&](int s0n, int bufn) {
        if (wIb < 4) {
            // waves 0-3: K rows s_rel [wIb*8, wIb*8+8)  (1 KB each, 2 x 512B instr)
#pragma unroll
            for (int j = 0; j < 2; ++j) {
                int off = wIb * 8 * 64 + (j * 32 + lane) * 8;        // halfs
                unsigned lds = (unsigned)(uintptr_t)(smem + bufn * 2048 + off);
                unsigned long long g =
                    (unsigned long long)(uintptr_t)(kb + s0n * DK + off);
                asm volatile("global_load_async_to_lds_b128 %0, %1, off"
                             :: "v"(lds), "v"(g) : "memory");
            }
        } else {
            // waves 4-7: V^T rows dk [wv*16, wv*16+16), 64B rows (4 lanes/row)
            int wv = wIb - 4;
#pragma unroll
            for (int j = 0; j < 2; ++j) {
                int row = wv * 16 + j * 8 + (lane >> 2);
                int c8  = (lane & 3) * 8;                            // halfs
                unsigned lds =
                    (unsigned)(uintptr_t)(smem + 4096 + bufn * 2048 + row * 32 + c8);
                unsigned long long g =
                    (unsigned long long)(uintptr_t)(vb + row * SEQ + s0n + c8);
                asm volatile("global_load_async_to_lds_b128 %0, %1, off"
                             :: "v"(lds), "v"(g) : "memory");
            }
        }
    };

    const v16h aq0 = load_a_frag(qb, DK, l16, half);
    const v16h aq1 = load_a_frag(qb + 32, DK, l16, half);

    v8f o[4] = {};
    float m[8], l[8];
#pragma unroll
    for (int r = 0; r < 8; ++r) { m[r] = -1e30f; l[r] = 0.0f; }

    const float sc = 0.125f * 1.44269504088896f;   // 1/sqrt(Dk) * log2(e)

    int buf = 0;
    issue_loads(0, 0);                              // prologue prefetch

    for (int s0 = 0; s0 < SEQ; s0 += 32) {
        // my prefetch into `buf` landed; everyone is done reading `buf^1`
        asm volatile("s_wait_asynccnt 0x0" ::: "memory");
        __syncthreads();
        if (s0 + 32 < SEQ) issue_loads(s0 + 32, buf ^ 1);

        const _Float16* kL = smem + buf * 2048;            // 32 x 64, ld=64
        const _Float16* vL = smem + 4096 + buf * 2048;     // 64 x 32, ld=32

        v16h bk00 = load_b_frag(kL,            DK, l16, half);
        v16h bk01 = load_b_frag(kL + 32,       DK, l16, half);
        v16h bk10 = load_b_frag(kL + 16 * DK,      DK, l16, half);
        v16h bk11 = load_b_frag(kL + 16 * DK + 32, DK, l16, half);

        v8f c0 = {}, c1 = {};
        c0 = wmma16(aq0, bk00, c0);
        c0 = wmma16(aq1, bk01, c0);
        c1 = wmma16(aq0, bk10, c1);
        c1 = wmma16(aq1, bk11, c1);

        float p0[8], p1[8], al[8];
#pragma unroll
        for (int r = 0; r < 8; ++r) {
            p0[r] = c0[r] * sc;
            p1[r] = c1[r] * sc;
            float t = fmaxf(p0[r], p1[r]);
            t = fmaxf(t, __shfl_xor(t, 1, 32));
            t = fmaxf(t, __shfl_xor(t, 2, 32));
            t = fmaxf(t, __shfl_xor(t, 4, 32));
            t = fmaxf(t, __shfl_xor(t, 8, 32));
            float mn = fmaxf(m[r], t);
            al[r] = exp2f(m[r] - mn);
            p0[r] = exp2f(p0[r] - mn);
            p1[r] = exp2f(p1[r] - mn);
            float rs = p0[r] + p1[r];
            rs += __shfl_xor(rs, 1, 32);
            rs += __shfl_xor(rs, 2, 32);
            rs += __shfl_xor(rs, 4, 32);
            rs += __shfl_xor(rs, 8, 32);
            l[r] = l[r] * al[r] + rs;
            m[r] = mn;
        }
#pragma unroll
        for (int j = 0; j < 4; ++j)
#pragma unroll
            for (int r = 0; r < 8; ++r) o[j][r] *= al[r];

        // C-layout probs -> LDS (row-major 16x32 f16) -> reload in A layout
#pragma unroll
        for (int r = 0; r < 8; ++r) {
            int row = r + 8 * half;
            pl[row * 32 + l16]      = (_Float16)p0[r];
            pl[row * 32 + 16 + l16] = (_Float16)p1[r];
        }
        asm volatile("s_wait_dscnt 0" ::: "memory");
        union { v16h v; v8h h[2]; } pa;
        pa.h[0] = *(const v8h*)(pl + l16 * 32 + 8 * half);
        pa.h[1] = *(const v8h*)(pl + l16 * 32 + 16 + 8 * half);

#pragma unroll
        for (int j = 0; j < 4; ++j) {
            v16h bv_ = load_b_frag(vL + j * 16 * 32, 32, l16, half);
            o[j] = wmma16(pa.v, bv_, o[j]);
        }
        buf ^= 1;
    }

    const int bI = bh >> 3, hI = bh & 7;
#pragma unroll
    for (int j = 0; j < 4; ++j) {
#pragma unroll
        for (int r = 0; r < 8; ++r) {
            float inv = 1.0f / l[r];
            int tok = bI * SEQ + q0 + r + 8 * half;
            int e   = hI * 64 + j * 16 + l16;
            oh[tok * D_MODEL + e] = (_Float16)(o[j][r] * inv);
        }
    }
}

// ---------------------------------------------------------------------------
// Output projection: out = O @ Wo^T + bo (fp32 result)
// ---------------------------------------------------------------------------
__global__ void out_proj(const _Float16* __restrict__ ohx, const _Float16* __restrict__ woh,
                         const float* __restrict__ bo, float* __restrict__ out) {
    const int lane = threadIdx.x & 31;
    const int l16 = lane & 15, half = lane >> 4;
    const int w  = blockIdx.x * (blockDim.x >> 5) + (threadIdx.x >> 5);
    const int tm = w % 256;
    const int tn = w / 256;

    const _Float16* xa = ohx + tm * 16 * D_MODEL;
    v8f c[4] = {};
    for (int kc = 0; kc < D_MODEL / 32; ++kc) {
        v16h a = load_a_frag(xa + kc * 32, D_MODEL, l16, half);
#pragma unroll
        for (int j = 0; j < 4; ++j) {
            v16h b = load_b_frag(woh + (tn * 64 + j * 16) * D_MODEL + kc * 32, D_MODEL, l16, half);
            c[j] = wmma16(a, b, c[j]);
        }
    }
    const int t0 = tm * 16;
#pragma unroll
    for (int j = 0; j < 4; ++j) {
        int e = tn * 64 + j * 16 + l16;
        float bb = bo[e];
#pragma unroll
        for (int r = 0; r < 8; ++r) {
            int tok = t0 + r + 8 * half;
            out[tok * D_MODEL + e] = c[j][r] + bb;
        }
    }
}

// ---------------------------------------------------------------------------
extern "C" void kernel_launch(void* const* d_in, const int* in_sizes, int n_in,
                              void* d_out, int out_size, void* d_ws, size_t ws_size,
                              hipStream_t stream) {
    const float* x  = (const float*)d_in[0];
    const float* Wq = (const float*)d_in[1];
    const float* bq = (const float*)d_in[2];
    const float* Wk = (const float*)d_in[3];
    const float* bk = (const float*)d_in[4];
    const float* Wv = (const float*)d_in[5];
    const float* bv = (const float*)d_in[6];
    const float* Wo = (const float*)d_in[7];
    const float* bo = (const float*)d_in[8];
    float* out = (float*)d_out;

    // workspace carve (halfs)
    const size_t NX = (size_t)NTOK * D_MODEL;        // 2M
    const size_t NW = (size_t)D_MODEL * D_MODEL;     // 256K
    _Float16* ws  = (_Float16*)d_ws;
    _Float16* xh  = ws;
    _Float16* wqh = xh + NX;
    _Float16* wkh = wqh + NW;
    _Float16* wvh = wkh + NW;
    _Float16* woh = wvh + NW;
    _Float16* qh  = woh + NW;
    _Float16* kh  = qh + NX;
    _Float16* vth = kh + NX;
    _Float16* oh  = vth + NX;

    cvt_f32_f16<<<1024, 256, 0, stream>>>(x,  xh,  (int)NX);
    cvt_f32_f16<<<256,  256, 0, stream>>>(Wq, wqh, (int)NW);
    cvt_f32_f16<<<256,  256, 0, stream>>>(Wk, wkh, (int)NW);
    cvt_f32_f16<<<256,  256, 0, stream>>>(Wv, wvh, (int)NW);
    cvt_f32_f16<<<256,  256, 0, stream>>>(Wo, woh, (int)NW);

    // 3 matrices x 2048 tiles = 6144 waves, 8 waves/block
    qkv_proj<<<768, 256, 0, stream>>>(xh, wqh, wkh, wvh, bq, bk, bv, qh, kh, vth);

    // B*H*(S/16) = 2048 waves, 8 per block sharing one (b,h) K/V stream
    attn_kernel<<<256, 256, 0, stream>>>(qh, kh, vth, oh);

    // 2048 waves
    out_proj<<<256, 256, 0, stream>>>(oh, woh, bo, out);
}
